// SupConLoss_26594437497345
// MI455X (gfx1250) — matrix operand order repaired and checked by reading, not compiled, not run on hardware
//
#include <hip/hip_runtime.h>
#include <math.h>

typedef __attribute__((ext_vector_type(16))) __bf16        v16bf;
typedef __attribute__((ext_vector_type(8)))  float         v8f;
typedef __attribute__((ext_vector_type(4)))  unsigned int  u32x4;

constexpr int   B_SZ = 4096;
constexpr int   D_SZ = 1024;
constexpr float TINV = 1.0f / 0.07f;      // 1/temperature
constexpr float M0   = 1.0f / 0.07f;      // fixed softmax shift (== diagonal sim value)
constexpr float EPS  = 1e-8f;

union FragBF { u32x4 u[2]; v16bf v; };

__device__ __forceinline__ unsigned short f2bf(float f) {
  unsigned int u = __float_as_uint(f);
  unsigned int r = 0x7fffu + ((u >> 16) & 1u);   // round-to-nearest-even
  return (unsigned short)((u + r) >> 16);
}

__device__ __forceinline__ void top5_insert(float t[5], float cand) {
#pragma unroll
  for (int k = 0; k < 5; ++k) {
    float big = fmaxf(t[k], cand);
    cand      = fminf(t[k], cand);
    t[k]      = big;
  }
}

// ---------------------------------------------------------------------------
// Kernel 1: row-wise L2 normalize (f32 -> bf16 into workspace). One block/row.
// Block (0,0) also zeroes the loss accumulator.
// ---------------------------------------------------------------------------
__global__ __launch_bounds__(256) void normalize_cast_kernel(
    const float* __restrict__ features, unsigned short* __restrict__ fbf,
    float* __restrict__ out) {
  const int row  = blockIdx.x;
  const int tid  = threadIdx.x;
  const int lane = tid & 31;
  const int wave = tid >> 5;
  if (row == 0 && tid == 0) out[0] = 0.0f;

  const float4 x = ((const float4*)(features + (size_t)row * D_SZ))[tid];
  float ss = x.x * x.x + x.y * x.y + x.z * x.z + x.w * x.w;
#pragma unroll
  for (int off = 16; off >= 1; off >>= 1) ss += __shfl_xor(ss, off);

  __shared__ float warp_ss[8];
  if (lane == 0) warp_ss[wave] = ss;
  __syncthreads();
  float tot = 0.0f;
#pragma unroll
  for (int w = 0; w < 8; ++w) tot += warp_ss[w];

  const float scale = 1.0f / fmaxf(sqrtf(tot), 1e-12f);  // matches F.normalize

  uint2 o;
  o.x = (unsigned)f2bf(x.x * scale) | ((unsigned)f2bf(x.y * scale) << 16);
  o.y = (unsigned)f2bf(x.z * scale) | ((unsigned)f2bf(x.w * scale) << 16);
  *(uint2*)(fbf + (size_t)row * D_SZ + tid * 4) = o;
}

// ---------------------------------------------------------------------------
// Kernel 2: fused sim-GEMM (bf16 WMMA, f32 acc) + online SupCon statistics.
// 16-row strip per workgroup; 8 waves x 16 passes x 2 column tiles (A reused).
// ---------------------------------------------------------------------------
__global__ __launch_bounds__(256) void supcon_fused_kernel(
    const unsigned short* __restrict__ fbf, const int* __restrict__ labels,
    float* __restrict__ out) {
  const int tid  = threadIdx.x;
  const int lane = tid & 31;
  const int wave = tid >> 5;
  const int hi   = lane >> 4;   // half-wave id
  const int lo   = lane & 15;
  const int m0   = blockIdx.x * 16;

  // labels of the 8 rows this thread accumulates (C/D layout: M = v + 8*hi)
  int lblI[8];
#pragma unroll
  for (int v = 0; v < 8; ++v) lblI[v] = labels[m0 + v + 8 * hi];

  // per-row online state (fixed-shift softmax; raw-valued top-5 of negatives)
  float m_[8], p_[8], q_[8], cnt_[8], t_[8][5];
#pragma unroll
  for (int v = 0; v < 8; ++v) {
    m_[v] = -__builtin_inff(); p_[v] = 0.f; q_[v] = 0.f; cnt_[v] = 0.f;
#pragma unroll
    for (int k = 0; k < 5; ++k) t_[v][k] = -__builtin_inff();
  }

  // A fragment base: 16-bit A 16x32 layout -> lane holds row (m0+lo),
  // K = k0 + 16*hi .. +16 contiguous (matches B).
  const unsigned short* aRow = fbf + (size_t)(m0 + lo) * D_SZ + 16 * hi;

  for (int tt = 0; tt < 16; ++tt) {
    const int n0 = (wave * 32 + tt * 2) * 16;  // first of two adjacent tiles
    const int n1 = n0 + 16;
    const int lblJ0 = labels[n0 + lo];
    const int lblJ1 = labels[n1 + lo];
    // B fragment: 32x16 layout -> lane holds column (n+lo) = feature row,
    // K = k0 + 16*hi .. +16 contiguous.
    const unsigned short* bRow0 = fbf + (size_t)(n0 + lo) * D_SZ + 16 * hi;
    const unsigned short* bRow1 = fbf + (size_t)(n1 + lo) * D_SZ + 16 * hi;

    v8f c0 = {};
    v8f c1 = {};
#pragma unroll 4
    for (int k0 = 0; k0 < D_SZ; k0 += 32) {
      FragBF a, b0, b1;
      a.u[0]  = *(const u32x4*)(aRow + k0);
      a.u[1]  = *(const u32x4*)(aRow + k0 + 8);
      b0.u[0] = *(const u32x4*)(bRow0 + k0);
      b0.u[1] = *(const u32x4*)(bRow0 + k0 + 8);
      b1.u[0] = *(const u32x4*)(bRow1 + k0);
      b1.u[1] = *(const u32x4*)(bRow1 + k0 + 8);
      c0 = __builtin_amdgcn_wmma_f32_16x16x32_bf16(false, a.v, false, b0.v,
                                                   (short)0, c0, false, false);
      c1 = __builtin_amdgcn_wmma_f32_16x16x32_bf16(false, a.v, false, b1.v,
                                                   (short)0, c1, false, false);
    }

    // fused epilogue: clip/temperature + online stats, for both tiles
#pragma unroll
    for (int half = 0; half < 2; ++half) {
      const int   nB   = half ? n1 : n0;
      const int   lblJ = half ? lblJ1 : lblJ0;
#pragma unroll
      for (int v = 0; v < 8; ++v) {
        const float dot  = half ? c1[v] : c0[v];
        const int   i    = m0 + v + 8 * hi;
        const float s    = fminf(fmaxf(dot, -10.f), 10.f) * TINV;
        const bool  same = (lblI[v] == lblJ);
        const bool  diag = (i == nB + lo);
        m_[v] = fmaxf(m_[v], s);
        const float x = __expf(s - M0);          // never overflows: s <= ~M0
        p_[v]  += (same && !diag) ? x : 0.f;     // positives (excl. diagonal)
        q_[v]  += same ? 0.f : x;                // true negatives
        cnt_[v] += same ? 1.f : 0.f;             // same-label count (incl. diag)
        top5_insert(t_[v], same ? -__builtin_inff() : s);
      }
    }
  }

  // butterfly reduce across each 16-lane half (rows stay within a half)
#pragma unroll
  for (int off = 1; off <= 8; off <<= 1) {
#pragma unroll
    for (int v = 0; v < 8; ++v) {
      const float om = __shfl_xor(m_[v], off);
      const float op = __shfl_xor(p_[v], off);
      const float oq = __shfl_xor(q_[v], off);
      const float oc = __shfl_xor(cnt_[v], off);
      float o0 = __shfl_xor(t_[v][0], off);
      float o1 = __shfl_xor(t_[v][1], off);
      float o2 = __shfl_xor(t_[v][2], off);
      float o3 = __shfl_xor(t_[v][3], off);
      float o4 = __shfl_xor(t_[v][4], off);
      m_[v] = fmaxf(m_[v], om); p_[v] += op; q_[v] += oq; cnt_[v] += oc;
      top5_insert(t_[v], o0); top5_insert(t_[v], o1); top5_insert(t_[v], o2);
      top5_insert(t_[v], o3); top5_insert(t_[v], o4);
    }
  }

  __shared__ float red[16][8][10];
  if (lo == 0) {
#pragma unroll
    for (int v = 0; v < 8; ++v) {
      float* d = red[v + 8 * hi][wave];
      d[0] = m_[v]; d[1] = p_[v]; d[2] = q_[v]; d[3] = cnt_[v];
      d[4] = t_[v][0]; d[5] = t_[v][1]; d[6] = t_[v][2];
      d[7] = t_[v][3]; d[8] = t_[v][4];
    }
  }
  __syncthreads();

  if (tid < 16) {  // one thread finalizes each of the 16 rows
    float m = -__builtin_inff(), p = 0.f, q = 0.f, cf = 0.f;
    float t[5] = {-__builtin_inff(), -__builtin_inff(), -__builtin_inff(),
                  -__builtin_inff(), -__builtin_inff()};
#pragma unroll
    for (int w = 0; w < 8; ++w) {
      const float* d = red[tid][w];
      m = fmaxf(m, d[0]); p += d[1]; q += d[2]; cf += d[3];
      top5_insert(t, d[4]); top5_insert(t, d[5]); top5_insert(t, d[6]);
      top5_insert(t, d[7]); top5_insert(t, d[8]);
    }
    // correct the fixed shift to the true row max
    const float sc = __expf(M0 - m);
    p *= sc; q *= sc;
    const int nSame  = (int)(cf + 0.5f);
    const int numNeg = B_SZ - nSame;
    // masked (same-label) slots contribute exact 0.0 to the top-k pool
    const int nz = nSame < 5 ? nSame : 5;
    for (int z = 0; z < nz; ++z) top5_insert(t, 0.0f);
    const int   ak   = numNeg < 5 ? numNeg : 5;
    const float eneg = q + cf * __expf(-m);  // exp(neg_scores - m).sum()

    float eh[5], S = 0.f;
#pragma unroll
    for (int k = 0; k < 5; ++k) {
      const float tk = (k < ak) ? t[k] : 0.0f;   // where(col < actual_k, ., 0)
      eh[k] = __expf(tk - m);
      S += eh[k];
    }
    float rowLoss = 0.f;
#pragma unroll
    for (int k = 0; k < 5; ++k) {
      const float denom = p + S + eneg - eh[k];
      rowLoss -= __logf(p / (denom + EPS) + EPS);
    }
    atomicAdd(out, rowLoss * (1.0f / (B_SZ * 5.0f)));
  }
}

// ---------------------------------------------------------------------------
extern "C" void kernel_launch(void* const* d_in, const int* in_sizes, int n_in,
                              void* d_out, int out_size, void* d_ws, size_t ws_size,
                              hipStream_t stream) {
  (void)in_sizes; (void)n_in; (void)out_size; (void)ws_size;
  const float* features = (const float*)d_in[0];
  const int*   labels   = (const int*)d_in[1];
  float*       out      = (float*)d_out;
  unsigned short* fbf   = (unsigned short*)d_ws;  // 4096*1024 bf16 = 8 MB

  normalize_cast_kernel<<<B_SZ, 256, 0, stream>>>(features, fbf, out);
  supcon_fused_kernel<<<B_SZ / 16, 256, 0, stream>>>(fbf, labels, out);
}
